// EdgeDecoder_52218212384936
// MI455X (gfx1250) — matrix operand order repaired and checked by reading, not compiled, not run on hardware
//
#include <hip/hip_runtime.h>

typedef __bf16 bf16_t;
typedef __attribute__((ext_vector_type(16))) __bf16 v16bf;
typedef __attribute__((ext_vector_type(8)))  float  v8f;

#define HID   128            // hidden size
#define KDIM  256            // 2*HID
#define MPW   2              // M-tiles (16 edges each) per wave
#define TM    (8 * 16 * MPW) // 256 edges per block (8 waves)

__device__ __forceinline__ void split_bf16(float x, bf16_t& hi, bf16_t& lo) {
    hi = (bf16_t)x;                    // RNE f32->bf16
    lo = (bf16_t)(x - (float)hi);      // residual (lowers to v_fma_mix)
}

// ---------------------------------------------------------------------------
// Prep: W1 [128,256] f32  ->  64 B-fragments (ntile*8+kstep), hi/lo bf16.
// Fragment layout (wave32 WMMA B, 32x16 K x N):
//   lane L holds column n = ntile*16 + (L&15),
//   K range kstep*32 + (L<16 ? 0..15 : 16..31), ascending, 2 per VGPR.
// Storage: frag f, lane L -> 16 bf16 at [f*512 + L*16].
// ---------------------------------------------------------------------------
__global__ void prep_w1(const float* __restrict__ W1,
                        bf16_t* __restrict__ Bh, bf16_t* __restrict__ Bl) {
    int tid   = blockIdx.x * blockDim.x + threadIdx.x;   // 0..2047
    int f     = tid >> 5;
    int L     = tid & 31;
    int ntile = f >> 3;
    int kstep = f & 7;
    int n     = ntile * 16 + (L & 15);
    int kbase = kstep * 32 + ((L < 16) ? 0 : 16);
    int o     = f * 512 + L * 16;
    #pragma unroll
    for (int j = 0; j < 16; ++j) {
        float w = W1[n * KDIM + kbase + j];
        bf16_t h, l;
        split_bf16(w, h, l);
        Bh[o + j] = h;
        Bl[o + j] = l;
    }
}

// ---------------------------------------------------------------------------
// Main kernel: block = 256 threads (8 wave32s), TM = 256 edges.
// Wave w owns 2 M-tiles (32 edges) and all 128 N columns, so each B fragment
// load is amortized over 6 WMMAs.
// ---------------------------------------------------------------------------
__global__ __launch_bounds__(256) void edge_decoder(
    const float* __restrict__ z_src, const float* __restrict__ z_dst,
    const int*   __restrict__ eli,                 // [2, E] int32
    const bf16_t* __restrict__ Bh, const bf16_t* __restrict__ Bl,
    const float* __restrict__ b1, const float* __restrict__ W2,
    const float* __restrict__ b2,
    float* __restrict__ out, int E)
{
    const int lane  = threadIdx.x & 31;
    const int wave  = threadIdx.x >> 5;
    const int half  = (lane >> 4) & 1;             // 0: lanes 0-15, 1: lanes 16-31
    const int ln    = lane & 15;
    const int ebase = blockIdx.x * TM + wave * (16 * MPW);

    // Per-M-tile gathered row pointers. Lanes L and L+16 share row m = L&15
    // and load complementary K halves of it.
    const float* rowS[MPW];
    const float* rowD[MPW];
    #pragma unroll
    for (int mt = 0; mt < MPW; ++mt) {
        int e  = ebase + mt * 16 + ln;
        int ec = (e < E) ? e : 0;
        rowS[mt] = z_src + (size_t)eli[ec] * HID;
        rowD[mt] = z_dst + (size_t)eli[E + ec] * HID;
    }

    v8f acc[MPW][8];
    #pragma unroll
    for (int mt = 0; mt < MPW; ++mt)
        #pragma unroll
        for (int t = 0; t < 8; ++t)
            #pragma unroll
            for (int i = 0; i < 8; ++i) acc[mt][t][i] = 0.0f;

    #pragma unroll
    for (int kstep = 0; kstep < 8; ++kstep) {
        const int kb = (kstep & 3) * 32 + (half ? 16 : 0);

        // Load + split-convert the A fragment of each M-tile.
        // A layout (matches B frags): lanes 0-15 hold k = kb+{0..15};
        // lanes 16-31 the same +16 (kb already includes it).
        v16bf ah[MPW], al[MPW];
        #pragma unroll
        for (int mt = 0; mt < MPW; ++mt) {
            const float* row = (kstep < 4) ? rowS[mt] : rowD[mt];
            float4 a0 = *(const float4*)(row + kb);
            float4 a1 = *(const float4*)(row + kb + 4);
            float4 a2 = *(const float4*)(row + kb + 8);
            float4 a3 = *(const float4*)(row + kb + 12);
            float fa[16] = { a0.x, a0.y, a0.z, a0.w, a1.x, a1.y, a1.z, a1.w,
                             a2.x, a2.y, a2.z, a2.w, a3.x, a3.y, a3.z, a3.w };
            #pragma unroll
            for (int i = 0; i < 16; ++i) {
                bf16_t h, l;
                split_bf16(fa[i], h, l);
                ah[mt][i] = h;
                al[mt][i] = l;
            }
        }

        #pragma unroll
        for (int nt = 0; nt < 8; ++nt) {
            const int fr = nt * 8 + kstep;
            v16bf bh = *(const v16bf*)(Bh + fr * 512 + lane * 16);
            v16bf bl = *(const v16bf*)(Bl + fr * 512 + lane * 16);
            #pragma unroll
            for (int mt = 0; mt < MPW; ++mt) {
                // D = Ah*Bh + Ah*Bl + Al*Bh + C  (fp32-accurate split product)
                acc[mt][nt] = __builtin_amdgcn_wmma_f32_16x16x32_bf16(
                    false, ah[mt], false, bh, (short)0, acc[mt][nt], false, false);
                acc[mt][nt] = __builtin_amdgcn_wmma_f32_16x16x32_bf16(
                    false, ah[mt], false, bl, (short)0, acc[mt][nt], false, false);
                acc[mt][nt] = __builtin_amdgcn_wmma_f32_16x16x32_bf16(
                    false, al[mt], false, bh, (short)0, acc[mt][nt], false, false);
            }
        }
    }

    // ---- Layer 2: out[e] = b2 + sum_n W2[n] * relu(h[e,n] + b1[n]) ----
    // C layout: lane -> N column (ln), VGPR v -> M row (v or v+8).
    float b1v[8], w2v[8];
    #pragma unroll
    for (int nt = 0; nt < 8; ++nt) {
        int n = nt * 16 + ln;
        b1v[nt] = b1[n];
        w2v[nt] = W2[n];
    }
    const float bias2 = b2[0];

    #pragma unroll
    for (int mt = 0; mt < MPW; ++mt) {
        #pragma unroll
        for (int v = 0; v < 8; ++v) {
            float s = 0.0f;
            #pragma unroll
            for (int nt = 0; nt < 8; ++nt) {
                float h = acc[mt][nt][v] + b1v[nt];
                h = fmaxf(h, 0.0f);
                s = fmaf(h, w2v[nt], s);
            }
            // butterfly reduce over the 16 lanes of this half (sum over n)
            s += __shfl_xor(s, 1, 32);
            s += __shfl_xor(s, 2, 32);
            s += __shfl_xor(s, 4, 32);
            s += __shfl_xor(s, 8, 32);
            if (ln == 0) {
                int m  = half ? (v + 8) : v;
                int eo = ebase + mt * 16 + m;
                if (eo < E) out[eo] = s + bias2;
            }
        }
    }
}

// ---------------------------------------------------------------------------
extern "C" void kernel_launch(void* const* d_in, const int* in_sizes, int n_in,
                              void* d_out, int out_size, void* d_ws, size_t ws_size,
                              hipStream_t stream) {
    const float* z_src = (const float*)d_in[0];
    const float* z_dst = (const float*)d_in[1];
    const int*   eli   = (const int*)d_in[2];   // [2,E] int32
    const float* W1    = (const float*)d_in[3];
    const float* b1    = (const float*)d_in[4];
    const float* W2    = (const float*)d_in[5];
    const float* b2    = (const float*)d_in[6];
    float* out = (float*)d_out;

    const int E = in_sizes[2] / 2;

    bf16_t* Bh = (bf16_t*)d_ws;            // 64 frags * 512 bf16 = 64 KB
    bf16_t* Bl = Bh + 64 * 512;            // another 64 KB

    prep_w1<<<8, 256, 0, stream>>>(W1, Bh, Bl);

    const int blocks = (E + TM - 1) / TM;
    edge_decoder<<<blocks, 256, 0, stream>>>(z_src, z_dst, eli, Bh, Bl,
                                             b1, W2, b2, out, E);
}
